// InfoNCEWithAdj_43404939493421
// MI455X (gfx1250) — compile-verified, matrix operand order checked by reading, not run on hardware
//
#include <hip/hip_runtime.h>
#include <hip/hip_bf16.h>

typedef __attribute__((ext_vector_type(16))) __bf16        v16bf;
typedef __attribute__((ext_vector_type(8)))  float         v8f;
typedef __attribute__((ext_vector_type(4)))  unsigned int  v4u;
typedef __attribute__((ext_vector_type(4)))  float         v4f;
typedef __attribute__((ext_vector_type(4)))  unsigned short v4us;

#define N_ROWS 8192
#define DIM 128
#define SPLITS 8
#define COLS_PER_SPLIT (N_ROWS / SPLITS)      // 1024
#define TILES_PER_SPLIT (COLS_PER_SPLIT / 16) // 64
#define INV_T 10.0f

__device__ __forceinline__ unsigned short f32_to_bf16_rne(float f) {
  unsigned int u = __float_as_uint(f);
  u += 0x7FFFu + ((u >> 16) & 1u);
  return (unsigned short)(u >> 16);
}

// ---------------------------------------------------------------------------
// Kernel 1: L2-normalize rows of q and k, emit bf16. One wave (32 lanes) per
// row; each lane handles 4 contiguous floats (D=128 = 32*4).
// blockIdx < N_ROWS -> query, else key.
// ---------------------------------------------------------------------------
__global__ __launch_bounds__(32)
void normalize_bf16_kernel(const float* __restrict__ q,
                           const float* __restrict__ k,
                           unsigned short* __restrict__ qb,
                           unsigned short* __restrict__ kb) {
  int bid = blockIdx.x;
  const float* src;
  unsigned short* dst;
  int row;
  if (bid < N_ROWS) { src = q; dst = qb; row = bid; }
  else              { src = k; dst = kb; row = bid - N_ROWS; }

  int lane = threadIdx.x;                       // 0..31 (wave32)
  v4f x = *((const v4f*)(src + (size_t)row * DIM) + lane);
  float ss = x.x * x.x + x.y * x.y + x.z * x.z + x.w * x.w;
#pragma unroll
  for (int m = 16; m >= 1; m >>= 1) ss += __shfl_xor(ss, m, 32);

  float nrm = sqrtf(ss);
  float inv = 1.0f / fmaxf(nrm, 1e-12f);

  v4us o;
  o.x = f32_to_bf16_rne(x.x * inv);
  o.y = f32_to_bf16_rne(x.y * inv);
  o.z = f32_to_bf16_rne(x.z * inv);
  o.w = f32_to_bf16_rne(x.w * inv);
  *((v4us*)(dst + (size_t)row * DIM) + lane) = o;
}

// ---------------------------------------------------------------------------
// Kernel 2: streaming logits pass. Each wave handles one 16-row tile of q
// and one column split (1024 columns = 64 tiles of 16). Per tile:
//   4x v_wmma_f32_16x16x32_bf16 (K = 128), scale by 1/T, load matching
//   16x16 adj tile (non-temporal: adj is 256 MB single-use stream; keep the
//   4 MB of reused bf16 operands resident in L2/WGP$), accumulate
//   sumexp / sum(adj*x) / sum(adj).
// 16-bit A/B fragment layout (ISA 7.12.2): lane L holds row (L&15); lanes
// 0-15 take K-halves {0..7,16..23}, lanes 16-31 take {8..15,24..31} of each
// 32-wide K chunk. B shares the same per-lane pattern with lane = column,
// which for q*k^T means loading k rows verbatim (no transpose).
// C layout: reg r, lane L -> (m = r + 8*(L>>4), n = L&15).
// Inputs are L2-normalized so |logit| <= 1/T = 10: sumexp <= 8192*e^10,
// no overflow possible -> skip online-max tracking entirely.
// ---------------------------------------------------------------------------
__global__ __launch_bounds__(256)
void infonce_main_kernel(const unsigned short* __restrict__ qb,
                         const unsigned short* __restrict__ kb,
                         const float* __restrict__ adj,
                         float* __restrict__ p_se,
                         float* __restrict__ p_S,
                         float* __restrict__ p_A) {
  int wave = threadIdx.x >> 5;
  int lane = threadIdx.x & 31;
  int gw   = blockIdx.x * 8 + wave;             // 0 .. 4095
  int row_tile = gw >> 3;                       // 0 .. 511
  int split    = gw & (SPLITS - 1);             // 0 .. 7
  int row_base = row_tile * 16;
  int col_base = split * COLS_PER_SPLIT;

  int hi = lane >> 4;                           // half-wave select
  int lr = lane & 15;

  union Frag { v4u u[2]; v16bf v; };

  // A fragments: q rows, reused across the whole column loop.
  Frag a[4];
  const unsigned short* qrow = qb + (size_t)(row_base + lr) * DIM;
#pragma unroll
  for (int kk = 0; kk < 4; ++kk) {
    const unsigned short* p = qrow + kk * 32 + hi * 8;
    a[kk].u[0] = *(const v4u*)p;
    a[kk].u[1] = *(const v4u*)(p + 16);
  }

  float acc_se[8], acc_S[8], acc_A[8];
#pragma unroll
  for (int r = 0; r < 8; ++r) { acc_se[r] = 0.f; acc_S[r] = 0.f; acc_A[r] = 0.f; }

  const float* adj_base =
      adj + (size_t)(row_base + hi * 8) * N_ROWS + (size_t)col_base + lr;

  for (int ct = 0; ct < TILES_PER_SPLIT; ++ct) {
    const unsigned short* krow =
        kb + (size_t)(col_base + ct * 16 + lr) * DIM;

    v8f c = {};
#pragma unroll
    for (int kk = 0; kk < 4; ++kk) {
      Frag b;
      const unsigned short* p = krow + kk * 32 + hi * 8;
      b.u[0] = *(const v4u*)p;
      b.u[1] = *(const v4u*)(p + 16);
      c = __builtin_amdgcn_wmma_f32_16x16x32_bf16(
          /*neg_a=*/false, a[kk].v, /*neg_b=*/false, b.v,
          /*c_mod=*/(short)0, c, /*reuse_a=*/false, /*reuse_b=*/false);
    }

    const float* ap = adj_base + ct * 16;
#pragma unroll
    for (int r = 0; r < 8; ++r) {
      float x  = c[r] * INV_T;                  // |x| <= 10: exp cannot overflow
      float ad = __builtin_nontemporal_load(ap + (size_t)r * N_ROWS);
      acc_se[r] += __expf(x);
      acc_S[r]   = fmaf(ad, x, acc_S[r]);
      acc_A[r]  += ad;
    }
  }

  // Reduce across the 16 lanes holding each row (half-waves stay disjoint).
#pragma unroll
  for (int m = 1; m <= 8; m <<= 1) {
#pragma unroll
    for (int r = 0; r < 8; ++r) {
      acc_se[r] += __shfl_xor(acc_se[r], m, 32);
      acc_S[r]  += __shfl_xor(acc_S[r],  m, 32);
      acc_A[r]  += __shfl_xor(acc_A[r],  m, 32);
    }
  }

  if (lr == 0) {
#pragma unroll
    for (int r = 0; r < 8; ++r) {
      int row = row_base + hi * 8 + r;
      size_t idx = (size_t)split * N_ROWS + row;  // unique per (split,row)
      __builtin_nontemporal_store(acc_se[r], p_se + idx);
      __builtin_nontemporal_store(acc_S[r],  p_S  + idx);
      __builtin_nontemporal_store(acc_A[r],  p_A  + idx);
    }
  }
}

// ---------------------------------------------------------------------------
// Kernel 3: combine splits, pos_i = S_i - A_i * log(sumexp_i), reduce to
// out[0] = -mean(pos). Single block, deterministic tree reduction.
// ---------------------------------------------------------------------------
__global__ __launch_bounds__(256)
void infonce_finalize_kernel(const float* __restrict__ p_se,
                             const float* __restrict__ p_S,
                             const float* __restrict__ p_A,
                             float* __restrict__ out) {
  __shared__ float red[256];
  float acc = 0.f;
  for (int i = threadIdx.x; i < N_ROWS; i += 256) {
    float se = 0.f, S = 0.f, A = 0.f;
#pragma unroll
    for (int s = 0; s < SPLITS; ++s) {
      se += p_se[(size_t)s * N_ROWS + i];
      S  += p_S[(size_t)s * N_ROWS + i];
      A  += p_A[(size_t)s * N_ROWS + i];
    }
    acc += S - logf(se) * A;
  }
  red[threadIdx.x] = acc;
  __syncthreads();
  for (int s = 128; s > 0; s >>= 1) {
    if ((int)threadIdx.x < s) red[threadIdx.x] += red[threadIdx.x + s];
    __syncthreads();
  }
  if (threadIdx.x == 0) out[0] = -red[0] / (float)N_ROWS;
}

// ---------------------------------------------------------------------------
extern "C" void kernel_launch(void* const* d_in, const int* in_sizes, int n_in,
                              void* d_out, int out_size, void* d_ws, size_t ws_size,
                              hipStream_t stream) {
  const float* q   = (const float*)d_in[0];
  const float* k   = (const float*)d_in[1];
  const float* adj = (const float*)d_in[2];

  char* ws = (char*)d_ws;
  unsigned short* qb = (unsigned short*)ws;                         // 2 MB
  unsigned short* kb = (unsigned short*)(ws + (size_t)2 * 1024 * 1024);
  float* p_se = (float*)(ws + (size_t)4 * 1024 * 1024);             // 256 KB
  float* p_S  = p_se + (size_t)SPLITS * N_ROWS;                     // 256 KB
  float* p_A  = p_S  + (size_t)SPLITS * N_ROWS;                     // 256 KB

  normalize_bf16_kernel<<<2 * N_ROWS, 32, 0, stream>>>(q, k, qb, kb);

  int total_waves = (N_ROWS / 16) * SPLITS;                         // 4096
  infonce_main_kernel<<<total_waves / 8, 256, 0, stream>>>(
      qb, kb, adj, p_se, p_S, p_A);

  infonce_finalize_kernel<<<1, 256, 0, stream>>>(p_se, p_S, p_A, (float*)d_out);
}
